// CausalSelfAttention_14482629722829
// MI455X (gfx1250) — compile-verified, matrix operand order checked by reading, not run on hardware
//
#include <hip/hip_runtime.h>
#include <hip/hip_bf16.h>

typedef _Float16 f16;
typedef __attribute__((ext_vector_type(16))) _Float16 v16h;
typedef __attribute__((ext_vector_type(8)))  _Float16 h8;
typedef __attribute__((ext_vector_type(8)))  float    v8f;

#define CEMBD   2048
#define NHEAD   16
#define HDIM    128
#define BATCH   2
#define SEQ     2048
#define BT      (BATCH * SEQ)     // 4096
#define C3      (3 * CEMBD)       // 6144

// ---------------------------------------------------------------------------
// WMMA helper
// ---------------------------------------------------------------------------
__device__ __forceinline__ v8f wmma_f16(v16h a, v16h b, v8f c) {
  return __builtin_amdgcn_wmma_f32_16x16x32_f16(false, a, false, b, (short)0, c,
                                                false, false);
}

// Load a 16x32 f16 fragment with the WMMA K-dimension contiguous in memory.
// Lane r=lane&15, half=lane>>4; elements 0..7 -> K=k0+8*half.., 8..15 -> K=k0+16+8*half..
// => two 16-byte contiguous loads per lane (global_load_b128 or ds_load_b128).
__device__ __forceinline__ v16h load_frag16(const f16* __restrict__ base,
                                            int ld, int row0, int k0) {
  const int lane = threadIdx.x & 31;
  const int r    = lane & 15;
  const int half = lane >> 4;
  const f16* p = base + (size_t)(row0 + r) * (size_t)ld + (size_t)(k0 + half * 8);
  h8 lo = *(const h8*)(p);
  h8 hi = *(const h8*)(p + 16);
  return __builtin_shufflevector(lo, hi, 0, 1, 2, 3, 4, 5, 6, 7,
                                        8, 9, 10, 11, 12, 13, 14, 15);
}

// gfx1250 async global -> LDS copy (16B per lane), tracked by ASYNCcnt.
// VDST = per-lane LDS byte address, VADDR = per-lane 64-bit global address.
__device__ __forceinline__ void async_copy_b128(unsigned int lds_byte,
                                                const void* gptr) {
  asm volatile("global_load_async_to_lds_b128 %0, %1, off"
               :: "v"(lds_byte), "v"(gptr) : "memory");
}
// Async loads retire in order: waiting ASYNCcnt<=4 retires the previous tile
// (4 ops/thread) while the next tile's 4 ops remain in flight.
__device__ __forceinline__ void async_wait0() {
  asm volatile("s_wait_asynccnt 0x0" ::: "memory");
}
__device__ __forceinline__ void async_wait4() {
  asm volatile("s_wait_asynccnt 0x4" ::: "memory");
}

// ---------------------------------------------------------------------------
// Precision conversion / weight transpose kernels
// ---------------------------------------------------------------------------
__global__ void __launch_bounds__(256)
convert_f32_f16_kernel(const float* __restrict__ in, f16* __restrict__ out, int n) {
  int i = blockIdx.x * blockDim.x + threadIdx.x;
  if (i < n) out[i] = (f16)in[i];
}

// W: [K, N] row major  ->  Wt: [N, K] row major (f16)
__global__ void __launch_bounds__(256)
transpose_w_kernel(const float* __restrict__ W, f16* __restrict__ Wt, int K, int N) {
  size_t i = (size_t)blockIdx.x * blockDim.x + threadIdx.x;
  size_t total = (size_t)K * (size_t)N;
  if (i < total) {
    size_t nn = i / (size_t)K;
    size_t kk = i % (size_t)K;
    Wt[i] = (f16)W[kk * (size_t)N + nn];
  }
}

// ---------------------------------------------------------------------------
// QKV GEMM: block = 256 threads = 8 waves as 4(M) x 2(N); wave tile 32x64,
// block tile 128x128. 8 WMMAs per 6 fragment loads per wave per k-step.
// Epilogue scatters into Qb/Kb [B,H,T,D] and Vt [B,H,D,T] (f16).
// ---------------------------------------------------------------------------
__global__ void __launch_bounds__(256)
qkv_gemm_kernel(const f16* __restrict__ xh, const f16* __restrict__ WaT,
                const float* __restrict__ b_attn,
                f16* __restrict__ Qb, f16* __restrict__ Kb, f16* __restrict__ Vt) {
  const int wv   = threadIdx.x >> 5;
  const int lane = threadIdx.x & 31;
  const int mw   = wv & 3;
  const int nw   = wv >> 2;
  const int row0 = blockIdx.y * 128 + mw * 32;
  const int col0 = blockIdx.x * 128 + nw * 64;

  v8f acc[2][4] = {};
  for (int k0 = 0; k0 < CEMBD; k0 += 32) {
    // speculative prefetch of A eight k-steps ahead (dropped if useless)
    __builtin_prefetch((const void*)(xh + (size_t)(row0 + (lane & 15)) * CEMBD +
                                     k0 + 256), 0, 1);
    v16h a0 = load_frag16(xh, CEMBD, row0,      k0);
    v16h a1 = load_frag16(xh, CEMBD, row0 + 16, k0);
#pragma unroll
    for (int t = 0; t < 4; ++t) {
      v16h b = load_frag16(WaT, CEMBD, col0 + 16 * t, k0);
      acc[0][t] = wmma_f16(a0, b, acc[0][t]);
      acc[1][t] = wmma_f16(a1, b, acc[1][t]);
    }
  }

  const int n = lane & 15, half = lane >> 4;
#pragma unroll
  for (int t = 0; t < 4; ++t) {
    const int d    = col0 + 16 * t + n;     // qkv column 0..6143
    const float bi = b_attn[d];
    const int kind = d >> 11;               // 0=q 1=k 2=v
    const int c    = d & (CEMBD - 1);
    const int hd   = c >> 7;
    const int dd   = c & (HDIM - 1);
#pragma unroll
    for (int r = 0; r < 2; ++r) {
#pragma unroll
      for (int j = 0; j < 8; ++j) {
        const int m  = row0 + 16 * r + j + 8 * half;   // token 0..4095
        const int bb = m >> 11;
        const int tt = m & (SEQ - 1);
        const f16 val = (f16)(acc[r][t][j] + bi);
        const size_t bh = (size_t)(bb * NHEAD + hd);
        if (kind == 0)      Qb[(bh * SEQ + tt) * HDIM + dd] = val;
        else if (kind == 1) Kb[(bh * SEQ + tt) * HDIM + dd] = val;
        else                Vt[(bh * HDIM + dd) * SEQ + tt] = val;
      }
    }
  }
}

// ---------------------------------------------------------------------------
// Flash attention: one 8-wave block owns 128 consecutive queries of one (b,h).
// 32-key K/V tiles are staged in double-buffered LDS via
// global_load_async_to_lds_b128; copies for tile i+1 overlap compute of tile i
// (in-order ASYNCcnt retirement, s_wait_asynccnt 4). Each staged tile is
// consumed by all 8 waves: 8 score WMMAs + online softmax + 8 PV WMMAs each.
// Causal skip is wave-uniform so EXEC is all-ones around every WMMA.
// ---------------------------------------------------------------------------
__global__ void __launch_bounds__(256)
flash_attn_kernel(const f16* __restrict__ Qb, const f16* __restrict__ Kb,
                  const f16* __restrict__ Vt, f16* __restrict__ yh) {
  // LDS: K[2][32][128] (16KB) | V[2][128][32] (16KB) | P per wave [16][32] (8KB)
  __shared__ __align__(16) f16 lds[2 * 32 * 128 + 2 * 128 * 32 + 8 * 16 * 32];

  const int wv   = threadIdx.x >> 5;
  const int lane = threadIdx.x & 31;
  const int bh   = blockIdx.x >> 4;          // 0..31  (b*16 + head)
  const int qblk = blockIdx.x & 15;          // 128-query block within sequence
  const int base = qblk * 128;
  const int q0   = base + wv * 16;           // this wave's query tile

  const f16* Qm = Qb + (size_t)bh * SEQ * HDIM;
  const f16* Km = Kb + (size_t)bh * SEQ * HDIM;
  const f16* Vm = Vt + (size_t)bh * HDIM * SEQ;
  f16* Psh = lds + 2 * 32 * 128 + 2 * 128 * 32 + wv * (16 * 32);

  const unsigned int lds_base = (unsigned int)(uintptr_t)lds;  // LDS byte offset
  const unsigned int k_base   = lds_base;
  const unsigned int v_base   = lds_base + 2 * 32 * 128 * 2;

  // stage K tile (32x128) + V tile (128x32) for tile index `it` (4 ops/thread)
  auto stage_tile = [&](int it) {
    const int n0 = it * 32;
    const unsigned int kb = k_base + (unsigned)(it & 1) * (32 * 128 * 2);
    const unsigned int vb = v_base + (unsigned)(it & 1) * (128 * 32 * 2);
#pragma unroll
    for (int i = 0; i < 2; ++i) {
      int idx = (int)threadIdx.x + 256 * i;  // 0..511, 16B each
      int krow = idx >> 4, kseg = idx & 15;  // K: 16 x b128 per row
      async_copy_b128(kb + idx * 16,
                      Km + (size_t)(n0 + krow) * HDIM + kseg * 8);
      int vrow = idx >> 2, vseg = idx & 3;   // V: 4 x b128 per row
      async_copy_b128(vb + idx * 16,
                      Vm + (size_t)vrow * SEQ + n0 + vseg * 8);
    }
  };

  v16h qf[4];
#pragma unroll
  for (int c = 0; c < 4; ++c) qf[c] = load_frag16(Qm, HDIM, q0, 32 * c);

  v8f   acc[8] = {};
  float mrow[8], lrow[8];
#pragma unroll
  for (int j = 0; j < 8; ++j) { mrow[j] = -INFINITY; lrow[j] = 0.0f; }

  const float sc   = 0.08838834764831845f;   // 1/sqrt(128)
  const int   n    = lane & 15;
  const int   half = lane >> 4;
  const int   ntiles = (base + 128) / 32;    // block's causal frontier

  stage_tile(0);
  for (int it = 0; it < ntiles; ++it) {
    const int n0 = it * 32;
    if (it + 1 < ntiles) { stage_tile(it + 1); async_wait4(); }
    else                 { async_wait0(); }
    __syncthreads();                         // tile `it` visible to all waves

    if (n0 < q0 + 16) {                      // wave-uniform causal skip
      const f16* Ksh = lds + (it & 1) * (32 * 128);
      const f16* Vsh = lds + 2 * 32 * 128 + (it & 1) * (128 * 32);
      // ---- S = Q . K^T ---------------------------------------------------
      v8f s0 = {}, s1 = {};
#pragma unroll
      for (int c = 0; c < 4; ++c) {
        v16h kf0 = load_frag16(Ksh, HDIM, 0,  32 * c);
        v16h kf1 = load_frag16(Ksh, HDIM, 16, 32 * c);
        s0 = wmma_f16(qf[c], kf0, s0);
        s1 = wmma_f16(qf[c], kf1, s1);
      }
      // ---- causal mask + online softmax ----------------------------------
#pragma unroll
      for (int j = 0; j < 8; ++j) {
        const int m = q0 + j + 8 * half;
        float z0 = (n0 + n      <= m) ? s0[j] * sc : -INFINITY;
        float z1 = (n0 + 16 + n <= m) ? s1[j] * sc : -INFINITY;
        float rmax = fmaxf(z0, z1);
#pragma unroll
        for (int off = 1; off < 16; off <<= 1)
          rmax = fmaxf(rmax, __shfl_xor(rmax, off, 16));
        const float mnew = fmaxf(mrow[j], rmax);
        const float p0 = __expf(z0 - mnew);
        const float p1 = __expf(z1 - mnew);
        float rsum = p0 + p1;
#pragma unroll
        for (int off = 1; off < 16; off <<= 1)
          rsum += __shfl_xor(rsum, off, 16);
        const float f = __expf(mrow[j] - mnew);
        lrow[j] = lrow[j] * f + rsum;
        mrow[j] = mnew;
#pragma unroll
        for (int t = 0; t < 8; ++t) acc[t][j] *= f;
        Psh[(j + 8 * half) * 32 + n]      = (f16)p0;
        Psh[(j + 8 * half) * 32 + 16 + n] = (f16)p1;
      }
      // ---- O += P . V ----------------------------------------------------
      v16h pf = load_frag16(Psh, 32, 0, 0);
#pragma unroll
      for (int t = 0; t < 8; ++t) {
        v16h vf = load_frag16(Vsh, 32, 16 * t, 0);
        acc[t] = wmma_f16(pf, vf, acc[t]);
      }
    }
    __syncthreads();   // all reads of this buffer done before it is re-staged
  }

  // ---- normalize, write y as [B*T, C] f16 --------------------------------
  const int bb = bh >> 4, hd = bh & 15;
#pragma unroll
  for (int t = 0; t < 8; ++t) {
#pragma unroll
    for (int j = 0; j < 8; ++j) {
      const int m = q0 + j + 8 * half;
      const float v = acc[t][j] / lrow[j];
      yh[((size_t)(bb * SEQ + m)) * CEMBD + hd * HDIM + 16 * t + n] = (f16)v;
    }
  }
}

// ---------------------------------------------------------------------------
// Output projection: same 32x64 wave tiling, f32 output + bias.
// ---------------------------------------------------------------------------
__global__ void __launch_bounds__(256)
proj_gemm_kernel(const f16* __restrict__ yh, const f16* __restrict__ WpT,
                 const float* __restrict__ b_proj, float* __restrict__ out) {
  const int wv   = threadIdx.x >> 5;
  const int lane = threadIdx.x & 31;
  const int mw   = wv & 3;
  const int nw   = wv >> 2;
  const int row0 = blockIdx.y * 128 + mw * 32;
  const int col0 = blockIdx.x * 128 + nw * 64;

  v8f acc[2][4] = {};
  for (int k0 = 0; k0 < CEMBD; k0 += 32) {
    __builtin_prefetch((const void*)(yh + (size_t)(row0 + (lane & 15)) * CEMBD +
                                     k0 + 256), 0, 1);
    v16h a0 = load_frag16(yh, CEMBD, row0,      k0);
    v16h a1 = load_frag16(yh, CEMBD, row0 + 16, k0);
#pragma unroll
    for (int t = 0; t < 4; ++t) {
      v16h b = load_frag16(WpT, CEMBD, col0 + 16 * t, k0);
      acc[0][t] = wmma_f16(a0, b, acc[0][t]);
      acc[1][t] = wmma_f16(a1, b, acc[1][t]);
    }
  }

  const int n = lane & 15, half = lane >> 4;
#pragma unroll
  for (int t = 0; t < 4; ++t) {
    const int d    = col0 + 16 * t + n;
    const float bi = b_proj[d];
#pragma unroll
    for (int r = 0; r < 2; ++r) {
#pragma unroll
      for (int j = 0; j < 8; ++j) {
        const int m = row0 + 16 * r + j + 8 * half;
        out[(size_t)m * CEMBD + d] = acc[r][t][j] + bi;
      }
    }
  }
}

// ---------------------------------------------------------------------------
// Launch
// ---------------------------------------------------------------------------
extern "C" void kernel_launch(void* const* d_in, const int* in_sizes, int n_in,
                              void* d_out, int out_size, void* d_ws, size_t ws_size,
                              hipStream_t stream) {
  (void)in_sizes; (void)n_in; (void)out_size; (void)ws_size;
  const float* x      = (const float*)d_in[0];
  const float* W_attn = (const float*)d_in[1];
  const float* b_attn = (const float*)d_in[2];
  const float* W_proj = (const float*)d_in[3];
  const float* b_proj = (const float*)d_in[4];
  float* out = (float*)d_out;

  char* ws = (char*)d_ws;
  size_t off = 0;
  auto alloc = [&](size_t bytes) -> void* {
    void* p = ws + off;
    off += (bytes + 255) & ~(size_t)255;
    return p;
  };
  f16* xh  = (f16*)alloc((size_t)BT * CEMBD * sizeof(f16));
  f16* WaT = (f16*)alloc((size_t)C3 * CEMBD * sizeof(f16));
  f16* WpT = (f16*)alloc((size_t)CEMBD * CEMBD * sizeof(f16));
  f16* Qb  = (f16*)alloc((size_t)BATCH * NHEAD * SEQ * HDIM * sizeof(f16));
  f16* Kb  = (f16*)alloc((size_t)BATCH * NHEAD * SEQ * HDIM * sizeof(f16));
  f16* Vt  = (f16*)alloc((size_t)BATCH * NHEAD * HDIM * SEQ * sizeof(f16));
  f16* yh  = (f16*)alloc((size_t)BT * CEMBD * sizeof(f16));

  {
    int nel = BT * CEMBD;
    convert_f32_f16_kernel<<<(nel + 255) / 256, 256, 0, stream>>>(x, xh, nel);
  }
  {
    size_t tot = (size_t)C3 * CEMBD;
    transpose_w_kernel<<<(int)((tot + 255) / 256), 256, 0, stream>>>(W_attn, WaT,
                                                                     CEMBD, C3);
  }
  {
    size_t tot = (size_t)CEMBD * CEMBD;
    transpose_w_kernel<<<(int)((tot + 255) / 256), 256, 0, stream>>>(W_proj, WpT,
                                                                     CEMBD, CEMBD);
  }

  qkv_gemm_kernel<<<dim3(C3 / 128, BT / 128), 256, 0, stream>>>(xh, WaT, b_attn,
                                                                Qb, Kb, Vt);

  // one block per (b,h, 128-query chunk): 32 * 16 = 512 blocks
  flash_attn_kernel<<<BATCH * NHEAD * (SEQ / 128), 256, 0, stream>>>(Qb, Kb, Vt,
                                                                     yh);

  proj_gemm_kernel<<<dim3(CEMBD / 128, BT / 128), 256, 0, stream>>>(yh, WpT,
                                                                    b_proj, out);
}
